// HeatmapMultiBox2dDecoder_15719580304028
// MI455X (gfx1250) — matrix operand order repaired and checked by reading, not compiled, not run on hardware
//
#include <hip/hip_runtime.h>
#include <hip/hip_bf16.h>
#include <stdint.h>

typedef __attribute__((ext_vector_type(16))) _Float16 v16h;
typedef __attribute__((ext_vector_type(8)))  float    v8f;

#define C_CLS   8
#define HH      16
#define WW      16
#define HW      256          // H*W
#define KTOP    32           // MAX_BOX_NUM
#define MCAND   256          // C_CLS * KTOP
#define THRESH  0.1f         // SUBBOX_SCORE_THRESH
#define IOU_T   0.5f         // NMS_IOU_THRESH
#define CLS_OFF 4096.0f      // CLASS_OFFSET

// generic pointer into LDS -> wave-relative LDS byte offset (low 32 bits of flat addr)
__device__ __forceinline__ unsigned lds_off_u32(const void* p) {
    return (unsigned)(uintptr_t)p;
}

// CDNA5 async DMA: global -> LDS, 16 bytes per lane, tracked by ASYNCcnt
__device__ __forceinline__ void async_ld_b128(unsigned ldsoff, const void* g) {
    unsigned long long ga = (unsigned long long)(uintptr_t)g;
    asm volatile("global_load_async_to_lds_b128 %0, %1, off"
                 :: "v"(ldsoff), "v"(ga) : "memory");
}

__global__ __launch_bounds__(256)
void hm_decode_nms_kernel(const float* __restrict__ rois,
                          const float* __restrict__ cls,
                          const float* __restrict__ reg,
                          float* __restrict__ out, int BN)
{
    __shared__ __align__(16) float s_cls[C_CLS * HW];   // 8 KB class heatmaps
    __shared__ __align__(16) float s_reg[4 * HW];       // 4 KB reg offsets
    __shared__ float    s_cbox[MCAND * 4];              // candidate boxes (class-major)
    __shared__ float    s_csc [MCAND];                  // candidate scores
    __shared__ float    s_ss  [MCAND];                  // sort keys
    __shared__ int      s_sp  [MCAND];                  // sort permutation
    __shared__ float    s_sb  [MCAND * 4];              // sorted class-offset boxes
    __shared__ float    s_sar [MCAND];                  // sorted areas
    __shared__ _Float16 s_kf  [MCAND];                  // keep flags (f16 for WMMA scan)
    __shared__ float    s_pref[MCAND];                  // WMMA row-prefix sums
    __shared__ float    s_ob  [KTOP * 4];
    __shared__ float    s_os  [KTOP];
    __shared__ int      s_ol  [KTOP];
    __shared__ int      s_keepflag;

    const int tid  = threadIdx.x;
    const int bn   = blockIdx.x;
    const int wave = tid >> 5;
    const int lane = tid & 31;

    // ---------- 1. async DMA of this ROI's cls/reg tiles into LDS ----------
    {
        const float* gc = cls + (size_t)bn * (C_CLS * HW);
        const float* gr = reg + (size_t)bn * (4 * HW);
        async_ld_b128(lds_off_u32(&s_cls[tid * 4]),         gc + tid * 4);
        async_ld_b128(lds_off_u32(&s_cls[(tid + 256) * 4]), gc + (tid + 256) * 4);
        async_ld_b128(lds_off_u32(&s_reg[tid * 4]),         gr + tid * 4);
        asm volatile("s_wait_asynccnt 0" ::: "memory");
    }
    __syncthreads();

    // ---------- ROI decode constants (uniform; redundant per thread) -------
    const float* r = rois + (size_t)bn * 4;
    const float x1r = r[0], y1r = r[1], x2r = r[2], y2r = r[3];
    const float cx  = (x1r + x2r) * 0.5f;
    const float cy  = (y1r + y2r) * 0.5f;
    const float rw  = (x2r - x1r);                 // ROI_WH_ZOOM_SCALE = (1,1)
    const float rh  = (y2r - y1r);
    const float rx1 = cx - rw * 0.5f;
    const float ry1 = cy - rh * 0.5f;
    const float bsw = rw / (float)WW;
    const float bsh = rh / (float)HH;

    // ---------- 2. per-class top-32 (wave w handles class w) ---------------
    {
        const int c = wave;
        unsigned taken = 0;                         // lane owns s = e*32+lane
        for (int kk = 0; kk < KTOP; ++kk) {
            float bs = -1.0f; int bi = 1 << 30;
            #pragma unroll
            for (int e = 0; e < 8; ++e) {
                if (!((taken >> e) & 1u)) {
                    int   s = e * 32 + lane;
                    float v = s_cls[c * HW + s];
                    v = (v > THRESH) ? v : 0.0f;    // masked scores like reference
                    if (v > bs || (v == bs && s < bi)) { bs = v; bi = s; }
                }
            }
            #pragma unroll
            for (int d = 16; d > 0; d >>= 1) {      // wave32 butterfly arg-max
                float ov = __shfl_xor(bs, d, 32);
                int   oi = __shfl_xor(bi, d, 32);
                if (ov > bs || (ov == bs && oi < bi)) { bs = ov; bi = oi; }
            }
            if ((bi & 31) == lane) taken |= 1u << (bi >> 5);   // owner retires slot
            if (lane == kk) {                       // lane kk writes candidate kk
                int m = c * KTOP + kk;
                s_csc[m] = bs;
                float bx1 = 0.f, by1 = 0.f, bx2 = 0.f, by2 = 0.f;
                if (bs > THRESH) {                  // mask applied to gathered boxes
                    int h = bi >> 4, w = bi & 15;
                    bx1 = bsw * (s_reg[0 * HW + bi] + (float)w + 0.5f) + rx1;
                    by1 = bsh * (s_reg[1 * HW + bi] + (float)h + 0.5f) + ry1;
                    bx2 = bsw * (s_reg[2 * HW + bi] + (float)w + 0.5f) + rx1;
                    by2 = bsh * (s_reg[3 * HW + bi] + (float)h + 0.5f) + ry1;
                }
                s_cbox[m * 4 + 0] = bx1; s_cbox[m * 4 + 1] = by1;
                s_cbox[m * 4 + 2] = bx2; s_cbox[m * 4 + 3] = by2;
            }
        }
    }
    __syncthreads();

    // ---------- 3. bitonic sort: score desc, index asc (== argsort(-s)) ----
    s_ss[tid] = s_csc[tid];
    s_sp[tid] = tid;
    __syncthreads();
    for (int k = 2; k <= MCAND; k <<= 1) {
        for (int j = k >> 1; j > 0; j >>= 1) {
            int ixj = tid ^ j;
            if (ixj > tid) {
                float a = s_ss[tid], b = s_ss[ixj];
                int  pa = s_sp[tid], pb = s_sp[ixj];
                bool inorder = (a > b) || (a == b && pa < pb);
                bool up = ((tid & k) == 0);
                if (up != inorder) {
                    s_ss[tid] = b; s_ss[ixj] = a;
                    s_sp[tid] = pb; s_sp[ixj] = pa;
                }
            }
            __syncthreads();
        }
    }

    // ---------- 4. class-offset boxes + IoU bit-columns --------------------
    const float msc  = s_ss[tid];
    const int   mp   = s_sp[tid];
    const int   mlab = mp >> 5;                     // label = m / KTOP
    const float coff = (float)mlab * CLS_OFF;
    const float ax1  = s_cbox[mp * 4 + 0] + coff;
    const float ay1  = s_cbox[mp * 4 + 1] + coff;
    const float ax2  = s_cbox[mp * 4 + 2] + coff;
    const float ay2  = s_cbox[mp * 4 + 3] + coff;
    const float aarea = fmaxf(ax2 - ax1, 0.f) * fmaxf(ay2 - ay1, 0.f);
    s_sb[tid * 4 + 0] = ax1; s_sb[tid * 4 + 1] = ay1;
    s_sb[tid * 4 + 2] = ax2; s_sb[tid * 4 + 3] = ay2;
    s_sar[tid] = aarea;
    __syncthreads();

    unsigned col[8] = {0, 0, 0, 0, 0, 0, 0, 0};     // col[i] = IoU(i, tid) > T
    for (int i = 0; i < MCAND; ++i) {               // broadcast LDS reads
        float bx1 = s_sb[i * 4 + 0], by1 = s_sb[i * 4 + 1];
        float bx2 = s_sb[i * 4 + 2], by2 = s_sb[i * 4 + 3];
        float iw = fmaxf(fminf(ax2, bx2) - fmaxf(ax1, bx1), 0.f);
        float ih = fmaxf(fminf(ay2, by2) - fmaxf(ay1, by1), 0.f);
        float inter = iw * ih;
        float uni   = aarea + s_sar[i] - inter;
        float iou   = inter / fmaxf(uni, 1e-9f);
        if (iou > IOU_T) col[i >> 5] |= 1u << (i & 31);
    }

    // ---------- 5. sequential greedy suppression (the true dependency) -----
    bool valid = (msc >= THRESH);
    bool sup   = false;
    for (int i = 0; i < MCAND; ++i) {
        if (tid == i) s_keepflag = (valid && !sup) ? 1 : 0;
        __syncthreads();
        if (s_keepflag && tid > i && ((col[i >> 5] >> (i & 31)) & 1u)) sup = true;
        __syncthreads();
    }
    const bool keep = valid && !sup;

    // ---------- 6. rank = cumsum(keep)-1 via WMMA scan (16x16 · upper-tri) --
    s_kf[tid] = (_Float16)(keep ? 1.0f : 0.0f);
    __syncthreads();
    if (tid < 32) {                                  // wave 0, EXEC all ones
        v16h A, Bm; v8f acc = {};
        const int  row = tid & 15;
        const bool hi  = tid >= 16;
        #pragma unroll
        for (int h = 0; h < 16; ++h) {
            // A (16x32 f16) layout: lanes 0-15 K={0..7,16..23}, lanes 16-31 K={8..15,24..31}
            int K = (h < 8 ? h : h + 8) + (hi ? 8 : 0);
            A[h] = (K < 16) ? s_kf[row * 16 + K] : (_Float16)0.0f;
            // B (32x16): lane holds column n=row; same K mapping as A: lanes 0-15 K={0..7,16..23}, lanes 16-31 K={8..15,24..31}
            int Kb = (h < 8 ? h : h + 8) + (hi ? 8 : 0);
            Bm[h] = (_Float16)((Kb < 16 && Kb <= row) ? 1.0f : 0.0f);
        }
        acc = __builtin_amdgcn_wmma_f32_16x16x32_f16(
                  false, A, false, Bm, (short)0, acc, false, false);
        // D (16x16 f32): lane = N + 16*(M>=8), vgpr v -> M = v (+8 for hi lanes)
        const int N = tid & 15, mb = hi ? 8 : 0;
        #pragma unroll
        for (int v = 0; v < 8; ++v) s_pref[(v + mb) * 16 + N] = acc[v];
    }
    __syncthreads();
    float inc  = s_pref[tid];                        // inclusive prefix within row
    float roff = 0.f;
    const int rrow = tid >> 4;
    for (int rr = 0; rr < rrow; ++rr) roff += s_pref[rr * 16 + 15];
    const int rank = (int)(inc + roff + 0.5f) - 1;

    // ---------- 7. stage outputs in LDS, then write -------------------------
    if (tid < KTOP) {
        s_os[tid] = 0.f; s_ol[tid] = -1;
        s_ob[tid * 4 + 0] = 0.f; s_ob[tid * 4 + 1] = 0.f;
        s_ob[tid * 4 + 2] = 0.f; s_ob[tid * 4 + 3] = 0.f;
    }
    __syncthreads();
    if (keep && rank >= 0 && rank < KTOP) {
        s_os[rank] = msc;
        s_ol[rank] = mlab;
        s_ob[rank * 4 + 0] = s_cbox[mp * 4 + 0];     // un-offset boxes, exact
        s_ob[rank * 4 + 1] = s_cbox[mp * 4 + 1];
        s_ob[rank * 4 + 2] = s_cbox[mp * 4 + 2];
        s_ob[rank * 4 + 3] = s_cbox[mp * 4 + 3];
    }
    __syncthreads();
    if (tid < KTOP) {
        const size_t g   = (size_t)bn * KTOP + tid;
        float* outb = out;                                        // (BN,K,4)
        float* outs = out + (size_t)BN * KTOP * 4;                // (BN,K)
        int*   outl = (int*)(out + (size_t)BN * KTOP * 5);        // (BN,K) int32
        outb[g * 4 + 0] = s_ob[tid * 4 + 0];
        outb[g * 4 + 1] = s_ob[tid * 4 + 1];
        outb[g * 4 + 2] = s_ob[tid * 4 + 2];
        outb[g * 4 + 3] = s_ob[tid * 4 + 3];
        outs[g] = s_os[tid];
        outl[g] = s_ol[tid];
    }
}

extern "C" void kernel_launch(void* const* d_in, const int* in_sizes, int n_in,
                              void* d_out, int out_size, void* d_ws, size_t ws_size,
                              hipStream_t stream) {
    (void)n_in; (void)out_size; (void)d_ws; (void)ws_size;
    const float* rois = (const float*)d_in[0];
    const float* cls  = (const float*)d_in[1];
    const float* reg  = (const float*)d_in[2];
    const int BN = in_sizes[1] / (C_CLS * HW);       // 2048 for the reference shapes
    hm_decode_nms_kernel<<<BN, 256, 0, stream>>>(rois, cls, reg, (float*)d_out, BN);
}
